// ContrastiveLoss_18253611008279
// MI455X (gfx1250) — matrix operand order repaired and checked by reading, not compiled, not run on hardware
//
#include <hip/hip_runtime.h>
#include <stdint.h>

// Problem constants (from reference): B=512, ZI=16, T=8, D=128
#define BN 512
#define ZI 16
#define TT 8
#define DD 128
#define CT (BN * TT)       // 4096 flattened text tokens
#define IMG_ROWS (BN * ZI) // 8192
#define TXT_ROWS (BN * TT) // 4096

typedef __attribute__((ext_vector_type(16))) __bf16 v16bf;
typedef __attribute__((ext_vector_type(8)))  float  v8f;

__device__ __forceinline__ uint16_t f32_to_bf16_rne(float f) {
    union { float f; uint32_t u; } v;
    v.f = f;
    uint32_t u = v.u;
    u += 0x7FFFu + ((u >> 16) & 1u);   // round to nearest even
    return (uint16_t)(u >> 16);
}

// ---------------------------------------------------------------------------
// Kernel 1: L2-normalize rows of length 128, emit bf16. One wave32 per row.
// ---------------------------------------------------------------------------
__global__ void __launch_bounds__(256)
normalize_bf16_kernel(const float* __restrict__ in, uint16_t* __restrict__ out,
                      int nrows) {
    const int wave = threadIdx.x >> 5;
    const int lane = threadIdx.x & 31;
    const int row  = blockIdx.x * (blockDim.x >> 5) + wave;
    if (row >= nrows) return;

    const float* r = in + (size_t)row * DD;
    float x0 = r[lane], x1 = r[lane + 32], x2 = r[lane + 64], x3 = r[lane + 96];
    float s = x0 * x0 + x1 * x1 + x2 * x2 + x3 * x3;
#pragma unroll
    for (int off = 16; off >= 1; off >>= 1) s += __shfl_xor(s, off, 32);
    const float scale = 1.0f / fmaxf(sqrtf(s), 1e-12f);

    uint16_t* o = out + (size_t)row * DD;
    o[lane]      = f32_to_bf16_rne(x0 * scale);
    o[lane + 32] = f32_to_bf16_rne(x1 * scale);
    o[lane + 64] = f32_to_bf16_rne(x2 * scale);
    o[lane + 96] = f32_to_bf16_rne(x3 * scale);
}

// ---------------------------------------------------------------------------
// Kernel 2: WMMA GEMM with fused max over the 16 img tokens.
//   A tile = 16 tokens of image bi (16 x 128 bf16), preloaded once into
//   registers (4 K-step fragments = 32 VGPRs) and reused across 4 output
//   tiles (64 ct columns) per wave -> 16 WMMAs per wave, A traffic cut 4x.
//   Grid = (512, 8), 8 waves/block; wave w covers ct group blockIdx.y*8+w.
// ---------------------------------------------------------------------------
__global__ void __launch_bounds__(256)
wmma_max_kernel(const uint16_t* __restrict__ imgn,
                const uint16_t* __restrict__ textn,
                float* __restrict__ M) {
    const int wave   = threadIdx.x >> 5;
    const int lane   = threadIdx.x & 31;
    const int bi     = blockIdx.x;
    const int ctbase = (blockIdx.y * 8 + wave) * 64;  // 64 columns per wave
    const int l15    = lane & 15;
    const int hi     = lane >> 4;  // 0: lanes 0-15, 1: lanes 16-31

    // A fragment (16-bit A 16x32 layout): lane half selects K {0..7,16..23}
    // vs {8..15,24..31} within each 32-wide K step.
    const uint16_t* arow = imgn + ((size_t)bi * ZI + l15) * DD + hi * 8;

    union bfrag { uint4 u[2]; v16bf v; };
    bfrag fa[4];
#pragma unroll
    for (int kk = 0; kk < 4; ++kk) {
        fa[kk].u[0] = *(const uint4*)(arow + kk * 32);       // VGPR0-3
        fa[kk].u[1] = *(const uint4*)(arow + kk * 32 + 16);  // VGPR4-7
    }

#pragma unroll
    for (int t = 0; t < 4; ++t) {
        const int ct0 = ctbase + t * 16;
        // B fragment: lane l15 holds column N=l15 (= textn row ct0+l15);
        // lanes 0-15 hold K {0..7,16..23}, lanes 16-31 hold K {8..15,24..31} of each step.
        const uint16_t* brow = textn + (size_t)(ct0 + l15) * DD + hi * 8;

        v8f acc = {};
#pragma unroll
        for (int kk = 0; kk < 4; ++kk) {
            bfrag fb;
            fb.u[0] = *(const uint4*)(brow + kk * 32);
            fb.u[1] = *(const uint4*)(brow + kk * 32 + 16);
            acc = __builtin_amdgcn_wmma_f32_16x16x32_bf16(
                /*neg_a=*/false, fa[kk].v, /*neg_b=*/false, fb.v,
                /*c_mod=*/(short)0, acc, /*reuse_a=*/false, /*reuse_b=*/false);
        }

        // Max over the M dimension of the 16x16 tile.
        // Lane l (<16) VGPR j = (M=j, N=l); lane l+16 VGPR j = (M=j+8, N=l).
        float m = acc[0];
#pragma unroll
        for (int j = 1; j < 8; ++j) m = fmaxf(m, acc[j]);
        m = fmaxf(m, __shfl_xor(m, 16, 32));
        if (lane < 16) M[(size_t)bi * CT + ct0 + lane] = m;
    }
}

// ---------------------------------------------------------------------------
// Kernel 3: per-bi log-sum-exp over all 4096 columns (deterministic tree).
// ---------------------------------------------------------------------------
__global__ void __launch_bounds__(256)
row_lse_kernel(const float* __restrict__ M, float* __restrict__ rowlog) {
    __shared__ float sh[256];
    const int bi = blockIdx.x;
    const float* r = M + (size_t)bi * CT;
    float s = 0.0f;
    for (int j = threadIdx.x; j < CT; j += 256) s += expf(r[j]);
    sh[threadIdx.x] = s;
    __syncthreads();
#pragma unroll
    for (int off = 128; off >= 1; off >>= 1) {
        if (threadIdx.x < off) sh[threadIdx.x] += sh[threadIdx.x + off];
        __syncthreads();
    }
    if (threadIdx.x == 0) rowlog[bi] = logf(sh[0]);
}

// ---------------------------------------------------------------------------
// Kernel 4: per-(bt,t) log-sum-exp over all 512 rows. Thread per column;
// coalesced row-wise sweep, fixed order => deterministic.
// ---------------------------------------------------------------------------
__global__ void __launch_bounds__(256)
col_lse_kernel(const float* __restrict__ M, float* __restrict__ collog) {
    const int ct = blockIdx.x * blockDim.x + threadIdx.x;
    if (ct >= CT) return;
    float s = 0.0f;
    for (int r = 0; r < BN; ++r) s += expf(M[(size_t)r * CT + ct]);
    collog[ct] = logf(s);
}

// ---------------------------------------------------------------------------
// Kernel 5: loss = sum_b rowlog[b] + sum_ct collog[ct] - (9/8) sum_{b,t} diag
// Single block, deterministic tree reduce.
// ---------------------------------------------------------------------------
__global__ void __launch_bounds__(256)
final_kernel(const float* __restrict__ M, const float* __restrict__ rowlog,
             const float* __restrict__ collog, float* __restrict__ out) {
    __shared__ float sh[256];
    float acc = 0.0f;
    for (int b = threadIdx.x; b < BN; b += 256) {
        float diag = 0.0f, csum = 0.0f;
        const float* mrow = M + (size_t)b * CT + (size_t)b * TT;
#pragma unroll
        for (int t = 0; t < TT; ++t) {
            diag += mrow[t];
            csum += collog[b * TT + t];
        }
        acc += rowlog[b] + csum - 1.125f * diag;  // 9/8 = 1 (t2i) + 1/8 (i2t mean)
    }
    sh[threadIdx.x] = acc;
    __syncthreads();
#pragma unroll
    for (int off = 128; off >= 1; off >>= 1) {
        if (threadIdx.x < off) sh[threadIdx.x] += sh[threadIdx.x + off];
        __syncthreads();
    }
    if (threadIdx.x == 0) out[0] = sh[0];
}

// ---------------------------------------------------------------------------
extern "C" void kernel_launch(void* const* d_in, const int* in_sizes, int n_in,
                              void* d_out, int out_size, void* d_ws, size_t ws_size,
                              hipStream_t stream) {
    const float* img  = (const float*)d_in[0];  // (512,16,128) f32
    const float* text = (const float*)d_in[1];  // (512, 8,128) f32
    float* out = (float*)d_out;                 // scalar loss

    uint8_t* ws = (uint8_t*)d_ws;
    uint16_t* imgn   = (uint16_t*)(ws + 0);          // 8192*128 bf16 = 2 MB
    uint16_t* textn  = (uint16_t*)(ws + 2097152);    // 4096*128 bf16 = 1 MB
    float*    M      = (float*)   (ws + 3145728);    // 512*4096 f32  = 8 MB
    float*    rowlog = (float*)   (ws + 11534336);   // 512 f32
    float*    collog = (float*)   (ws + 11536384);   // 4096 f32

    normalize_bf16_kernel<<<IMG_ROWS / 8, 256, 0, stream>>>(img, imgn, IMG_ROWS);
    normalize_bf16_kernel<<<TXT_ROWS / 8, 256, 0, stream>>>(text, textn, TXT_ROWS);

    dim3 g2(BN, 8);  // 8 waves/block, each wave: 4 tiles (64 cols), A in regs
    wmma_max_kernel<<<g2, 256, 0, stream>>>(imgn, textn, M);

    row_lse_kernel<<<BN, 256, 0, stream>>>(M, rowlog);
    col_lse_kernel<<<CT / 256, 256, 0, stream>>>(M, collog);
    final_kernel<<<1, 256, 0, stream>>>(M, rowlog, collog, out);
}
